// Model_31817117729573
// MI455X (gfx1250) — compile-verified, matrix operand order checked by reading, not run on hardware
//
#include <hip/hip_runtime.h>
#include <math.h>

// ---------------- constants (match reference) ----------------
#define NCn 50000
#define NTn 64
#define ECn 400000
#define ETn 2048
#define GCn 10000
#define GTn 16
#define NHn 256
#define EMBn 1024

// ---------------- bf16 helpers (bit-level, avoid scalar __bf16 ops) ----
__device__ __forceinline__ unsigned short f2bf(float f) {
  unsigned u = __float_as_uint(f);
  u += 0x7fffu + ((u >> 16) & 1u);          // round-to-nearest-even
  return (unsigned short)(u >> 16);
}
__device__ __forceinline__ float bf2f(unsigned short h) {
  return __uint_as_float(((unsigned)h) << 16);
}

// ---------------- WMMA fragment types ----------------
typedef __bf16 v16bf __attribute__((ext_vector_type(16)));
typedef float  v8f   __attribute__((ext_vector_type(8)));

union Frag { v16bf v; unsigned int u[8]; };
union Acc  { v8f   v; float        f[8]; };

// A-matrix 16x32 bf16 fragment per ISA 7.12.2:
//   lane: row = lane&15, half = lane>>4
//   vector element pair j (elems 2j,2j+1) holds K = k0 + kbase(j,half) .. +1
__device__ __forceinline__ void load_a_frag(Frag& fr, const unsigned short* sA,
                                            int stride, int row, int half, int k0) {
#pragma unroll
  for (int j = 0; j < 8; ++j) {
    int kbase = ((2 * j) & 7) + 8 * (((j >> 2) << 1) + half);
    fr.u[j] = *reinterpret_cast<const unsigned int*>(&sA[row * stride + k0 + kbase]);
  }
}

// ---------------- generic utility kernels ----------------
__global__ void __launch_bounds__(256) zero_kernel(float* p, long n) {
  long i = (long)blockIdx.x * blockDim.x + threadIdx.x;
  long s = (long)gridDim.x * blockDim.x;
  for (; i < n; i += s) p[i] = 0.f;
}

// Pack row-major f32 B[K,N] into per-wave WMMA B-fragment order:
// out[(((nt*(K/32)+kt)*32+lane)*16)+e] = bf16(B[(kt*32 + kloc(lane,e))*N + nt*16 + (lane&15)])
__global__ void __launch_bounds__(256) pack_b_kernel(const float* B, int K, int N,
                                                     unsigned short* out) {
  int nK = K >> 5;
  long total = (long)K * N;
  long i = (long)blockIdx.x * blockDim.x + threadIdx.x;
  long s = (long)gridDim.x * blockDim.x;
  for (; i < total; i += s) {
    int e = (int)(i & 15);
    int lane = (int)((i >> 4) & 31);
    long rem = i >> 9;
    int kt = (int)(rem % nK);
    int nt = (int)(rem / nK);
    int half = lane >> 4;
    int kloc = (e & 7) + 8 * (((e >> 3) << 1) + half);
    int k = kt * 32 + kloc;
    int n = nt * 16 + (lane & 15);
    out[i] = f2bf(B[(size_t)k * N + n]);
  }
}

// ---------------- WMMA GEMM: D = act(A1@B1 + A2@B2 + bias) ----------------
// A1: bf16 [M,K1]; A2: f32 or bf16 [M,K2]; B packed fragments.
// Epilogue: tanh (optional); write bf16 OR atomic group-scatter into f32.
template <int TPW>
__global__ void __launch_bounds__(256)
gemm2_kernel(int M, int N,
             const unsigned short* A1, int K1, const unsigned short* B1p,
             const float* A2f, const unsigned short* A2b, int K2,
             const unsigned short* B2p,
             const float* bias, int do_tanh,
             unsigned short* outB, float* outAcc, const int* gids) {
  __shared__ unsigned short sA1[16 * 256];
  __shared__ unsigned short sA2[16 * 256];
  const int tid = threadIdx.x;
  const int m0 = blockIdx.x * 16;

  { // stage A1 tile (row-major bf16) via dword copies
    const int tot = 16 * (K1 >> 1);
    const int kw = K1 >> 1;
    for (int idx = tid; idx < tot; idx += 256) {
      int r = idx / kw, c = idx % kw;
      reinterpret_cast<unsigned int*>(sA1)[r * kw + c] =
          reinterpret_cast<const unsigned int*>(A1 + (size_t)(m0 + r) * K1)[c];
    }
  }
  if (K2 > 0) {
    if (A2f) { // f32 source -> convert on stage
      const int tot = 16 * K2;
      for (int idx = tid; idx < tot; idx += 256) {
        int r = idx / K2, c = idx % K2;
        sA2[r * K2 + c] = f2bf(A2f[(size_t)(m0 + r) * K2 + c]);
      }
    } else {
      const int kw = K2 >> 1;
      const int tot = 16 * kw;
      for (int idx = tid; idx < tot; idx += 256) {
        int r = idx / kw, c = idx % kw;
        reinterpret_cast<unsigned int*>(sA2)[r * kw + c] =
            reinterpret_cast<const unsigned int*>(A2b + (size_t)(m0 + r) * K2)[c];
      }
    }
  }
  __syncthreads();

  const int lane = tid & 31, w = tid >> 5;
  const int row = lane & 15, half = lane >> 4;
  Acc acc[TPW];
#pragma unroll
  for (int j = 0; j < TPW; ++j)
#pragma unroll
    for (int k = 0; k < 8; ++k) acc[j].f[k] = 0.f;

  const Frag* b1 = reinterpret_cast<const Frag*>(B1p);
  const int nK1 = K1 >> 5;
  for (int kt = 0; kt < nK1; ++kt) {
    Frag a; load_a_frag(a, sA1, K1, row, half, kt * 32);
#pragma unroll
    for (int j = 0; j < TPW; ++j) {
      int nt = w * TPW + j;
      Frag b = b1[((size_t)nt * nK1 + kt) * 32 + lane];
      acc[j].v = __builtin_amdgcn_wmma_f32_16x16x32_bf16(
          false, a.v, false, b.v, (short)0, acc[j].v, false, false);
    }
  }
  if (K2 > 0) {
    const Frag* b2 = reinterpret_cast<const Frag*>(B2p);
    const int nK2 = K2 >> 5;
    for (int kt = 0; kt < nK2; ++kt) {
      Frag a; load_a_frag(a, sA2, K2, row, half, kt * 32);
#pragma unroll
      for (int j = 0; j < TPW; ++j) {
        int nt = w * TPW + j;
        Frag b = b2[((size_t)nt * nK2 + kt) * 32 + lane];
        acc[j].v = __builtin_amdgcn_wmma_f32_16x16x32_bf16(
            false, a.v, false, b.v, (short)0, acc[j].v, false, false);
      }
    }
  }
  // epilogue: C/D layout -> lane: col=n0+(lane&15); vgpr v: row = v + 8*half
#pragma unroll
  for (int j = 0; j < TPW; ++j) {
    int col = (w * TPW + j) * 16 + (lane & 15);
    float bv = bias ? bias[col] : 0.f;
#pragma unroll
    for (int v = 0; v < 8; ++v) {
      int r = m0 + v + 8 * half;
      float val = acc[j].f[v] + bv;
      if (do_tanh) val = tanhf(val);
      if (gids) atomicAdd(&outAcc[(size_t)gids[r] * N + col], val);
      else outB[(size_t)r * N + col] = f2bf(val);
    }
  }
}

// ------------- WMMA edge-message kernel (gather + GEMM + scatter) -------------
// msg = tanh(c[src]@WmTop + ce@WmBot + bm); atomicAdd into agg[dst]
__global__ void __launch_bounds__(256)
edge_msg_kernel(const unsigned short* Cb, const int* src, const int* dst,
                const float* cef, const float* ceW, const float* ceb,
                const unsigned short* Bp, const float* Wm, const float* bm,
                float* agg) {
  __shared__ unsigned short sA[16 * 256];
  __shared__ float sCE[16 * 8];
  __shared__ float sWB[8 * 256];
  const int tid = threadIdx.x;
  const int e0 = blockIdx.x * 16;

  for (int idx = tid; idx < 16 * 128; idx += 256) { // gather 16 source rows
    int r = idx >> 7, c = idx & 127;
    int s = src[e0 + r];
    reinterpret_cast<unsigned int*>(sA)[r * 128 + c] =
        reinterpret_cast<const unsigned int*>(Cb + (size_t)s * 256)[c];
  }
  if (tid < 128) { // per-edge edge-embedding: tanh(cedge@ceW + ceb)
    int r = tid >> 3, h = tid & 7;
    const float* x = cef + (size_t)(e0 + r) * 4;
    float a = ceb[h];
#pragma unroll
    for (int k = 0; k < 4; ++k) a += x[k] * ceW[k * 8 + h];
    sCE[r * 8 + h] = tanhf(a);
  }
  for (int idx = tid; idx < 8 * 256; idx += 256) sWB[idx] = Wm[256 * 256 + idx];
  __syncthreads();

  const int lane = tid & 31, w = tid >> 5;
  const int row = lane & 15, half = lane >> 4;
  Acc acc[2];
#pragma unroll
  for (int j = 0; j < 2; ++j)
#pragma unroll
    for (int k = 0; k < 8; ++k) acc[j].f[k] = 0.f;

  const Frag* bp = reinterpret_cast<const Frag*>(Bp);
  for (int kt = 0; kt < 8; ++kt) {
    Frag a; load_a_frag(a, sA, 256, row, half, kt * 32);
#pragma unroll
    for (int j = 0; j < 2; ++j) {
      int nt = w * 2 + j;
      Frag b = bp[((size_t)nt * 8 + kt) * 32 + lane];
      acc[j].v = __builtin_amdgcn_wmma_f32_16x16x32_bf16(
          false, a.v, false, b.v, (short)0, acc[j].v, false, false);
    }
  }
#pragma unroll
  for (int j = 0; j < 2; ++j) {
    int col = (w * 2 + j) * 16 + (lane & 15);
    float bv = bm[col];
#pragma unroll
    for (int v = 0; v < 8; ++v) {
      int r = v + 8 * half;
      float val = acc[j].f[v] + bv;
#pragma unroll
      for (int h = 0; h < 8; ++h) val += sCE[r * 8 + h] * sWB[h * 256 + col];
      val = tanhf(val);
      atomicAdd(&agg[(size_t)dst[e0 + r] * 256 + col], val);
    }
  }
}

// ---------------- small scalar kernels (t-side + embeds) ----------------
__global__ void __launch_bounds__(256)
c_embed_kernel(const float* cf, const int* ctypes, const float* ope,
               const float* W, const float* b, unsigned short* out) {
  __shared__ float x[32];
  const int m = blockIdx.x, tid = threadIdx.x;
  if (tid < 24) x[tid] = cf[(size_t)m * 24 + tid];
  else if (tid < 32) x[tid] = ope[(size_t)ctypes[m] * 8 + (tid - 24)];
  __syncthreads();
  float a = b[tid];
#pragma unroll
  for (int k = 0; k < 32; ++k) a += x[k] * W[k * 256 + tid];
  out[(size_t)m * 256 + tid] = f2bf(tanhf(a));
}

__global__ void __launch_bounds__(256)
t_embed_kernel(const float* tf, const float* W, const float* b, float* out) {
  __shared__ float x[16];
  const int m = blockIdx.x, tid = threadIdx.x;
  if (tid < 16) x[tid] = tf[(size_t)m * 16 + tid];
  __syncthreads();
  float a = b[tid];
#pragma unroll
  for (int k = 0; k < 16; ++k) a += x[k] * W[k * 256 + tid];
  out[(size_t)m * 256 + tid] = tanhf(a);
}

__global__ void __launch_bounds__(256)
t_edge_kernel(const float* T, const int* src, const int* dst,
              const float* tef, const float* teW, const float* teb,
              const float* Wm, const float* bm, float* tagg) {
  __shared__ float xs[256];
  __shared__ float teh[8];
  const int e = blockIdx.x, tid = threadIdx.x;
  xs[tid] = T[(size_t)src[e] * 256 + tid];
  if (tid < 8) {
    const float* x = tef + (size_t)e * 4;
    float a = teb[tid];
#pragma unroll
    for (int k = 0; k < 4; ++k) a += x[k] * teW[k * 8 + tid];
    teh[tid] = tanhf(a);
  }
  __syncthreads();
  float a = bm[tid];
  for (int k = 0; k < 256; ++k) a += xs[k] * Wm[k * 256 + tid];
#pragma unroll
  for (int h = 0; h < 8; ++h) a += teh[h] * Wm[(256 + h) * 256 + tid];
  atomicAdd(&tagg[(size_t)dst[e] * 256 + tid], tanhf(a));
}

// out = tanh(concat([X, Y]) @ W + b), rows of 256+256, also used for t-cross
__global__ void __launch_bounds__(256)
t_concat_dense_kernel(const float* X, const float* Y, const float* W,
                      const float* b, float* out) {
  __shared__ float xs[512];
  const int m = blockIdx.x, tid = threadIdx.x;
  xs[tid] = X[(size_t)m * 256 + tid];
  xs[256 + tid] = Y[(size_t)m * 256 + tid];
  __syncthreads();
  float a = b[tid];
  for (int k = 0; k < 512; ++k) a += xs[k] * W[k * 256 + tid];
  out[(size_t)m * 256 + tid] = tanhf(a);
}

// Z = Tn @ Wbot  ([64,256] @ [256,256])
__global__ void __launch_bounds__(256)
z_kernel(const float* Tn, const float* Wbot, float* Z) {
  __shared__ float xs[256];
  const int m = blockIdx.x, tid = threadIdx.x;
  xs[tid] = Tn[(size_t)m * 256 + tid];
  __syncthreads();
  float a = 0.f;
  for (int k = 0; k < 256; ++k) a += xs[k] * Wbot[k * 256 + tid];
  Z[(size_t)m * 256 + tid] = a;
}

// Q = P^T @ C  ([64,NC] @ [NC,256]); block = t-row, thread = column
__global__ void __launch_bounds__(256)
q_kernel(const float* P, const unsigned short* Cb, float* Q, int M) {
  const int t = blockIdx.x, col = threadIdx.x;
  float a = 0.f;
  for (int r = 0; r < M; ++r)
    a += P[(size_t)r * 64 + t] * bf2f(Cb[(size_t)r * 256 + col]);
  Q[(size_t)t * 256 + col] = a;
}

// tfin = T @ W + b, scattered by group ids into tgrp[16,1024]
__global__ void __launch_bounds__(256)
t_final_kernel(const float* T, const float* W, const float* b,
               const int* gids, float* tgrp) {
  __shared__ float xs[256];
  const int m = blockIdx.x, tid = threadIdx.x;
  xs[tid] = T[(size_t)m * 256 + tid];
  __syncthreads();
  const int g = gids[m];
#pragma unroll
  for (int j = 0; j < 4; ++j) {
    int col = tid + j * 256;
    float a = b[col];
    for (int k = 0; k < 256; ++k) a += xs[k] * W[(size_t)k * 1024 + col];
    atomicAdd(&tgrp[(size_t)g * 1024 + col], a);
  }
}

// out[GC,16] = cgrp @ tgrp^T
__global__ void __launch_bounds__(256)
out_kernel(const float* cg, const float* tg, float* out) {
  const int tid = threadIdx.x;
  const int r = blockIdx.x * 16 + (tid >> 4);
  const int c = tid & 15;
  float a = 0.f;
  for (int k = 0; k < 1024; ++k)
    a += cg[(size_t)r * 1024 + k] * tg[(size_t)c * 1024 + k];
  out[(size_t)r * 16 + c] = a;
}

// ---------------- host-side orchestration ----------------
extern "C" void kernel_launch(void* const* d_in, const int* in_sizes, int n_in,
                              void* d_out, int out_size, void* d_ws, size_t ws_size,
                              hipStream_t stream) {
  (void)in_sizes; (void)n_in; (void)out_size; (void)ws_size;
  const float* cfeats   = (const float*)d_in[0];
  const float* cedgef   = (const float*)d_in[1];
  const float* tfeats   = (const float*)d_in[2];
  const float* tedgef   = (const float*)d_in[3];
  const float* placement= (const float*)d_in[4];
  // d_in[5] runtime_stats: unused by reference
  const float* op_emb   = (const float*)d_in[6];
  const float* cn_W = (const float*)d_in[7];  const float* cn_b = (const float*)d_in[8];
  const float* ce_W = (const float*)d_in[9];  const float* ce_b = (const float*)d_in[10];
  const float* tn_W = (const float*)d_in[11]; const float* tn_b = (const float*)d_in[12];
  const float* te_W = (const float*)d_in[13]; const float* te_b = (const float*)d_in[14];
  const float* cgWm = (const float*)d_in[15]; const float* cgbm = (const float*)d_in[16];
  const float* cgWu = (const float*)d_in[17]; const float* cgbu = (const float*)d_in[18];
  const float* tgWm = (const float*)d_in[19]; const float* tgbm = (const float*)d_in[20];
  const float* tgWu = (const float*)d_in[21]; const float* tgbu = (const float*)d_in[22];
  const float* ccW  = (const float*)d_in[23]; const float* ccb  = (const float*)d_in[24];
  const float* tcW  = (const float*)d_in[25]; const float* tcb  = (const float*)d_in[26];
  const float* cfW  = (const float*)d_in[27]; const float* cfb  = (const float*)d_in[28];
  const float* tfW  = (const float*)d_in[29]; const float* tfb  = (const float*)d_in[30];
  const int* ctypes = (const int*)d_in[31];
  const int* c_src  = (const int*)d_in[32];   const int* c_dst  = (const int*)d_in[33];
  const int* t_src  = (const int*)d_in[34];   const int* t_dst  = (const int*)d_in[35];
  const int* cgid   = (const int*)d_in[36];   const int* tgid   = (const int*)d_in[37];
  float* out = (float*)d_out;

  // workspace carve-out (all offsets 256B aligned)
  unsigned char* base = (unsigned char*)d_ws;
  size_t off = 0;
  auto carve = [&](size_t bytes) -> void* {
    void* p = base + off;
    off = (off + bytes + 255) & ~(size_t)255;
    return p;
  };
  unsigned short* cA   = (unsigned short*)carve((size_t)NCn * 256 * 2);
  unsigned short* cB   = (unsigned short*)carve((size_t)NCn * 256 * 2);
  float* agg  = (float*)carve((size_t)NCn * 256 * 4);
  float* cgrp = (float*)carve((size_t)GCn * 1024 * 4);
  float* tgrp = (float*)carve((size_t)GTn * 1024 * 4);
  float* tcur = (float*)carve((size_t)NTn * 256 * 4);
  float* tnew = (float*)carve((size_t)NTn * 256 * 4);
  float* tagg = (float*)carve((size_t)NTn * 256 * 4);
  float* Qb   = (float*)carve((size_t)NTn * 256 * 4);
  float* Zf   = (float*)carve((size_t)NTn * 256 * 4);
  unsigned short* pkWm[2], *pkWuT[2], *pkWuB[2], *pkCC[2];
  for (int i = 0; i < 2; ++i) {
    pkWm[i]  = (unsigned short*)carve((size_t)256 * 256 * 2);
    pkWuT[i] = (unsigned short*)carve((size_t)256 * 256 * 2);
    pkWuB[i] = (unsigned short*)carve((size_t)256 * 256 * 2);
    pkCC[i]  = (unsigned short*)carve((size_t)256 * 256 * 2);
  }
  unsigned short* pkCF = (unsigned short*)carve((size_t)256 * 1024 * 2);
  unsigned short* pkZ  = (unsigned short*)carve((size_t)64 * 256 * 2);

  auto packB = [&](const float* B, int K, int N, unsigned short* o) {
    long total = (long)K * N;
    int blocks = (int)((total + 255) / 256); if (blocks > 4096) blocks = 4096;
    pack_b_kernel<<<blocks, 256, 0, stream>>>(B, K, N, o);
  };
  auto zeroF = [&](float* p, long n) {
    zero_kernel<<<1024, 256, 0, stream>>>(p, n);
  };

  // --- pack weights into WMMA fragment layout (once per launch) ---
  for (int i = 0; i < 2; ++i) {
    packB(cgWm + (size_t)i * 264 * 256, 256, 256, pkWm[i]);            // node part of Wmsg
    packB(cgWu + (size_t)i * 512 * 256, 256, 256, pkWuT[i]);           // Wupd top
    packB(cgWu + (size_t)i * 512 * 256 + 256 * 256, 256, 256, pkWuB[i]); // Wupd bottom
    packB(ccW  + (size_t)i * 512 * 256, 256, 256, pkCC[i]);            // cc_W top
  }
  packB(cfW, 256, 1024, pkCF);

  // --- initial embeddings ---
  c_embed_kernel<<<NCn, 256, 0, stream>>>(cfeats, ctypes, op_emb, cn_W, cn_b, cA);
  t_embed_kernel<<<NTn, 256, 0, stream>>>(tfeats, tn_W, tn_b, tcur);

  // --- two message-passing + cross rounds ---
  for (int i = 0; i < 2; ++i) {
    // C graph: edge messages -> agg (fused gather/GEMM/tanh/scatter)
    zeroF(agg, (long)NCn * 256);
    edge_msg_kernel<<<ECn / 16, 256, 0, stream>>>(
        cA, c_src, c_dst, cedgef, ce_W, ce_b, pkWm[i],
        cgWm + (size_t)i * 264 * 256, cgbm + (size_t)i * 256, agg);
    // c_new = tanh(c@WuT + agg@WuB + b)   [WMMA, A2 = f32 agg]
    gemm2_kernel<2><<<NCn / 16, 256, 0, stream>>>(
        NCn, 256, cA, 256, pkWuT[i], agg, nullptr, 256, pkWuB[i],
        cgbu + (size_t)i * 256, 1, cB, nullptr, nullptr);

    // T graph (tiny): messages, update
    zeroF(tagg, (long)NTn * 256);
    t_edge_kernel<<<ETn, 256, 0, stream>>>(
        tcur, t_src, t_dst, tedgef, te_W, te_b,
        tgWm + (size_t)i * 264 * 256, tgbm + (size_t)i * 256, tagg);
    t_concat_dense_kernel<<<NTn, 256, 0, stream>>>(
        tcur, tagg, tgWu + (size_t)i * 512 * 256, tgbu + (size_t)i * 256, tnew);

    // cross for C: (P@t_new)@Wbot == P@(t_new@Wbot) = P@Z
    z_kernel<<<NTn, 256, 0, stream>>>(tnew, ccW + (size_t)i * 512 * 256 + 256 * 256, Zf);
    packB(Zf, 64, 256, pkZ);
    // c = tanh(c_new@ccTop + P@Z + b)   [WMMA, A2 = f32 placement, K2=64]
    gemm2_kernel<2><<<NCn / 16, 256, 0, stream>>>(
        NCn, 256, cB, 256, pkCC[i], placement, nullptr, 64, pkZ,
        ccb + (size_t)i * 256, 1, cA, nullptr, nullptr);

    // cross for T: Q = P^T@c (reduction), then tiny dense
    q_kernel<<<NTn, 256, 0, stream>>>(placement, cA, Qb, NCn);
    t_concat_dense_kernel<<<NTn, 256, 0, stream>>>(
        tnew, Qb, tcW + (size_t)i * 512 * 256, tcb + (size_t)i * 256, tcur);
  }

  // --- final projection + group segment-sum (fused scatter) ---
  zeroF(cgrp, (long)GCn * 1024);
  gemm2_kernel<8><<<NCn / 16, 256, 0, stream>>>(
      NCn, 1024, cA, 256, pkCF, nullptr, nullptr, 0, nullptr,
      cfb, 0, nullptr, cgrp, cgid);
  zeroF(tgrp, (long)GTn * 1024);
  t_final_kernel<<<NTn, 256, 0, stream>>>(tcur, tfW, tfb, tgid, tgrp);

  // --- output GEMM [GC,1024] @ [1024,GT]^T ---
  out_kernel<<<GCn / 16, 256, 0, stream>>>(cgrp, tgrp, out);
}
